// Attention_KV_54065048322135
// MI455X (gfx1250) — matrix operand order repaired and checked by reading, not compiled, not run on hardware
//
#include <hip/hip_runtime.h>
#include <hip/hip_bf16.h>

// ---------------------------------------------------------------------------
// Multi-source attention for MI455X (gfx1250, wave32, WMMA + TDM).
//   Q = (hs[0] @ Wq + bq) * 0.125  -> bf16 [B,H,S,DH]   (score scale folded in)
//   K = hs @ Wk + bk               -> bf16 [N*B,H,S,DH]
//   V = hs @ Wv + bv               -> bf16 [N*B,H,S,DH]
//   out = sum_n softmax(Q K_n^T) V_n   (flash-attention, f32 accum)
// ---------------------------------------------------------------------------

typedef __attribute__((ext_vector_type(16))) __bf16 v16bf;
typedef __attribute__((ext_vector_type(8)))  __bf16 v8bf;
typedef __attribute__((ext_vector_type(8)))  float  v8f;
typedef __attribute__((ext_vector_type(4)))  unsigned int u32x4;
typedef __attribute__((ext_vector_type(8)))  int i32x8;
typedef __attribute__((ext_vector_type(4)))  int i32x4;

#define NSRC   5
#define BATCH  4
#define SLEN   1024
#define DMODEL 768
#define HEADS  12
#define DHEAD  64

#define WMMA_BF16(A, B, C) \
  __builtin_amdgcn_wmma_f32_16x16x32_bf16(false, (A), false, (B), (short)0, (C), false, false)

#if __has_builtin(__builtin_amdgcn_tensor_load_to_lds) && \
    __has_builtin(__builtin_amdgcn_s_wait_tensorcnt)
#define USE_TDM 1
#endif

__device__ __forceinline__ v8f vzero8() {
  v8f z;
#pragma unroll
  for (int i = 0; i < 8; ++i) z[i] = 0.0f;
  return z;
}

// A-matrix 16x32 bf16 fragment: lane m = row; element j -> K = half*16 + j
// (contiguous, matching the B fragments). Two contiguous 16B runs per lane.
__device__ __forceinline__ v16bf load_a_frag(const __bf16* row32, int half) {
  v8bf lo = *(const v8bf*)(row32 + half * 16);
  v8bf hi = *(const v8bf*)(row32 + half * 16 + 8);
  v16bf a;
#pragma unroll
  for (int j = 0; j < 8; ++j) { a[j] = lo[j]; a[8 + j] = hi[j]; }
  return a;
}

#ifdef USE_TDM
// TDM: DMA a [rows=32] x [cols=64] bf16 tile (row stride 64 elems) from global
// to LDS (packed row-major). Descriptor per ISA 8.3/8.4; 2D tensor, groups 2/3
// unused (zero). 6-arg builtin form (this toolchain): g0, g1, g2, g3, g4, cpol.
__device__ __forceinline__ void tdm_load_tile_32x64(const void* gptr,
                                                    unsigned lds_off) {
  unsigned long long ga = (unsigned long long)(uintptr_t)gptr;
  u32x4 g0;
  g0[0] = 1u;                                      // count=1, user mode
  g0[1] = lds_off;                                 // lds_addr (bytes)
  g0[2] = (unsigned)(ga & 0xffffffffull);          // global_addr[31:0]
  g0[3] = (unsigned)((ga >> 32) & 0x01ffffffull)   // global_addr[56:32]
          | 0x80000000u;                           // type=2 (bits 127:126)
  i32x8 g1;
  g1[0] = (int)(1u << 16);                         // data_size=1 (2 bytes)
  g1[1] = (int)((unsigned)DHEAD << 16);            // tensor_dim0=64 (bits 63:48)
  g1[2] = (int)((unsigned)SLEN << 16);             // tensor_dim1=1024 (bits 95:80)
  g1[3] = (int)((unsigned)DHEAD << 16);            // tile_dim0=64 (bits 127:112)
  g1[4] = 32;                                      // tile_dim1=32, tile_dim2=0
  g1[5] = DHEAD;                                   // tensor_dim0_stride=64
  g1[6] = 0;
  g1[7] = 0;
  i32x4 z4 = {0, 0, 0, 0};
  i32x8 z8 = {0, 0, 0, 0, 0, 0, 0, 0};
  __builtin_amdgcn_tensor_load_to_lds(g0, g1, z4, z4, z8, 0);
}
#endif

// ---------------------------------------------------------------------------
// Projection GEMM: Y[(seg*H + h)*S + s][DHEAD] = (X[row] @ W[:,h*64..] + b)*sc
// Block tile 64(M) x 64(N == one head), 4 waves, K-step 32.
// ---------------------------------------------------------------------------
__global__ __launch_bounds__(128)
void proj_gemm_kernel(const float* __restrict__ X, const float* __restrict__ W,
                      const float* __restrict__ bias, __bf16* __restrict__ Y,
                      float outScale) {
  __shared__ __bf16 Xs[64][32];   // A tile (row-major)
  __shared__ __bf16 Wts[64][32];  // B tile transposed: [col][k]

  const int tid  = threadIdx.x;
  const int wv   = tid >> 5;
  const int lane = tid & 31;
  const int m    = lane & 15;
  const int half = lane >> 4;
  const int rowBase = blockIdx.x * 64;
  const int h       = blockIdx.y;
  const int colBase = h * 64;

  v8f acc[4];
#pragma unroll
  for (int nt = 0; nt < 4; ++nt) acc[nt] = vzero8();

  for (int k0 = 0; k0 < DMODEL; k0 += 32) {
    __syncthreads();
    // X tile 64x32 f32 -> bf16 (4 float4 per thread)
#pragma unroll
    for (int i = 0; i < 4; ++i) {
      int e = (tid + i * 128) * 4;
      int r = e >> 5, c = e & 31;
      float4 x4 = *(const float4*)(X + (size_t)(rowBase + r) * DMODEL + k0 + c);
      Xs[r][c + 0] = (__bf16)x4.x;  Xs[r][c + 1] = (__bf16)x4.y;
      Xs[r][c + 2] = (__bf16)x4.z;  Xs[r][c + 3] = (__bf16)x4.w;
    }
    // W tile 32x64 f32 -> transposed bf16
#pragma unroll
    for (int i = 0; i < 4; ++i) {
      int e  = (tid + i * 128) * 4;
      int kr = e >> 6, c = e & 63;
      float4 w4 = *(const float4*)(W + (size_t)(k0 + kr) * DMODEL + colBase + c);
      Wts[c + 0][kr] = (__bf16)w4.x;  Wts[c + 1][kr] = (__bf16)w4.y;
      Wts[c + 2][kr] = (__bf16)w4.z;  Wts[c + 3][kr] = (__bf16)w4.w;
    }
    __syncthreads();

    v16bf a = load_a_frag(&Xs[wv * 16 + m][0], half);
#pragma unroll
    for (int nt = 0; nt < 4; ++nt) {
      // B fragment: lane column N = m; element j -> K = half*16 + j (contig).
      v16bf bfr = *(const v16bf*)&Wts[nt * 16 + m][half * 16];
      acc[nt] = WMMA_BF16(a, bfr, acc[nt]);
    }
  }

  // Epilogue: (+bias)*outScale, f32->bf16, layout [(seg)*H + h][S][DHEAD]
  const int seg = rowBase / SLEN;  // (n*B+b) or b; 64 | 1024 so uniform per blk
#pragma unroll
  for (int nt = 0; nt < 4; ++nt) {
    int d = nt * 16 + m;
    float bval = bias[colBase + d];
#pragma unroll
    for (int r = 0; r < 8; ++r) {
      int row = rowBase + wv * 16 + 8 * half + r;
      int s   = row - seg * SLEN;
      size_t idx = (((size_t)seg * HEADS + h) * SLEN + s) * DHEAD + d;
      Y[idx] = (__bf16)((acc[nt][r] + bval) * outScale);
    }
  }
}

// ---------------------------------------------------------------------------
// Flash attention: block = (qblock of 64, h, b); 4 waves x 16 q-rows.
// K tile via TDM DMA (wave 0) overlapped with manual V load+transpose.
// ---------------------------------------------------------------------------
__global__ __launch_bounds__(128)
void flash_attn_kernel(const __bf16* __restrict__ Qb, const __bf16* __restrict__ Kb,
                       const __bf16* __restrict__ Vb, float* __restrict__ Out) {
  __shared__ __bf16 Ks[32][64];       // K tile, row-major   (4 KB)
  __shared__ __bf16 Vts[64][32];      // V tile, transposed  (4 KB)
  __shared__ __bf16 Ps[4][16][32];    // per-wave P staging  (4 KB)

  const int tid  = threadIdx.x;
  const int wv   = tid >> 5;
  const int lane = tid & 31;
  const int m    = lane & 15;
  const int half = lane >> 4;
  const int qblk = blockIdx.x * 64;
  const int h    = blockIdx.y;
  const int b    = blockIdx.z;

#ifdef USE_TDM
  const unsigned ksOff = (unsigned)(uintptr_t)(void*)&Ks[0][0];
#endif

  // Q fragments (already pre-scaled by 0.125 in the projection).
  const __bf16* qrow =
      Qb + (((size_t)b * HEADS + h) * SLEN + qblk + wv * 16 + m) * DHEAD;
  const v16bf aq0 = load_a_frag(qrow, half);       // d = 0..31
  const v16bf aq1 = load_a_frag(qrow + 32, half);  // d = 32..63

  v8f outAcc[4];
#pragma unroll
  for (int nt = 0; nt < 4; ++nt) outAcc[nt] = vzero8();

  const float NEG_BIG = -3.0e38f;

  for (int n = 0; n < NSRC; ++n) {
    const size_t kvbase = ((((size_t)n * BATCH + b) * HEADS + h) * SLEN) * DHEAD;
    const __bf16* Ksrc = Kb + kvbase;
    const __bf16* Vsrc = Vb + kvbase;

    v8f ctx[4];
#pragma unroll
    for (int nt = 0; nt < 4; ++nt) ctx[nt] = vzero8();
    v8f mrow, lrow;
#pragma unroll
    for (int r = 0; r < 8; ++r) { mrow[r] = NEG_BIG; lrow[r] = 0.0f; }

    for (int k0 = 0; k0 < SLEN; k0 += 32) {
      __syncthreads();   // previous tile fully consumed by all waves

#ifdef USE_TDM
      if (wv == 0) {     // one DMA per block; EXEC-independent, TENSORcnt++
        tdm_load_tile_32x64(Ksrc + (size_t)k0 * DHEAD, ksOff);
      }
#else
#pragma unroll
      for (int i = 0; i < 2; ++i) {
        int e  = (tid + i * 128) * 8;
        int kr = e >> 6, c = e & 63;
        *(v8bf*)&Ks[kr][c] =
            *(const v8bf*)(Ksrc + (size_t)(k0 + kr) * DHEAD + c);
      }
#endif
      // V tile load + transpose into LDS (overlaps the TDM DMA).
#pragma unroll
      for (int i = 0; i < 2; ++i) {
        int e  = (tid + i * 128) * 8;
        int kr = e >> 6, c = e & 63;
        v8bf vvec = *(const v8bf*)(Vsrc + (size_t)(k0 + kr) * DHEAD + c);
#pragma unroll
        for (int j = 0; j < 8; ++j) Vts[c + j][kr] = vvec[j];
      }
      if (k0 + 32 < SLEN) {  // global_prefetch_b8 of next V tile
        __builtin_prefetch(Vsrc + (size_t)(k0 + 32) * DHEAD + tid * 16, 0, 0);
      }
#ifdef USE_TDM
      if (wv == 0) __builtin_amdgcn_s_wait_tensorcnt(0);
#endif
      __syncthreads();

      // scores: two 16x16 key sub-tiles, d accumulated through C chaining
      v8f s0 = vzero8(), s1 = vzero8();
      {
        v16bf bk00 = *(const v16bf*)&Ks[m][half * 16];
        v16bf bk01 = *(const v16bf*)&Ks[m][32 + half * 16];
        s0 = WMMA_BF16(aq0, bk00, s0);
        s0 = WMMA_BF16(aq1, bk01, s0);
        v16bf bk10 = *(const v16bf*)&Ks[16 + m][half * 16];
        v16bf bk11 = *(const v16bf*)&Ks[16 + m][32 + half * 16];
        s1 = WMMA_BF16(aq0, bk10, s1);
        s1 = WMMA_BF16(aq1, bk11, s1);
      }

      // online softmax per q-row (row = element r + 8*half; cols across lanes)
      v8f p0v, p1v, scl;
#pragma unroll
      for (int r = 0; r < 8; ++r) {
        float x0 = s0[r];
        float x1 = s1[r];
        float vmax = fmaxf(x0, x1);
        vmax = fmaxf(vmax, __shfl_xor(vmax, 1));
        vmax = fmaxf(vmax, __shfl_xor(vmax, 2));
        vmax = fmaxf(vmax, __shfl_xor(vmax, 4));
        vmax = fmaxf(vmax, __shfl_xor(vmax, 8));
        float mOld = mrow[r];
        float mNew = fmaxf(mOld, vmax);
        float sc   = __expf(mOld - mNew);
        float p0   = __expf(x0 - mNew);
        float p1   = __expf(x1 - mNew);
        float psum = p0 + p1;
        psum += __shfl_xor(psum, 1);
        psum += __shfl_xor(psum, 2);
        psum += __shfl_xor(psum, 4);
        psum += __shfl_xor(psum, 8);
        lrow[r] = lrow[r] * sc + psum;
        mrow[r] = mNew;
        scl[r]  = sc;
        p0v[r]  = p0;
        p1v[r]  = p1;
      }

      // Transpose P (C layout) -> A layout through per-wave LDS slice.
#pragma unroll
      for (int r = 0; r < 8; ++r) {
        Ps[wv][8 * half + r][m]      = (__bf16)p0v[r];
        Ps[wv][8 * half + r][16 + m] = (__bf16)p1v[r];
      }
      v16bf ap = load_a_frag(&Ps[wv][m][0], half);

      // ctx = ctx*scale + P @ V  (rescale folded into the C operand)
#pragma unroll
      for (int nt = 0; nt < 4; ++nt) {
        v16bf bvf = *(const v16bf*)&Vts[nt * 16 + m][half * 16];
        v8f c = ctx[nt];
#pragma unroll
        for (int r = 0; r < 8; ++r) c[r] *= scl[r];
        ctx[nt] = WMMA_BF16(ap, bvf, c);
      }
    }  // key tiles

    // finish source n: out += ctx * (1/l)
#pragma unroll
    for (int r = 0; r < 8; ++r) {
      float inv = 1.0f / lrow[r];
#pragma unroll
      for (int nt = 0; nt < 4; ++nt) outAcc[nt][r] += ctx[nt][r] * inv;
    }
  }  // sources

  // Store: Out[(b*S + q)*768 + h*64 + d]
#pragma unroll
  for (int nt = 0; nt < 4; ++nt) {
    int d = h * DHEAD + nt * 16 + m;
#pragma unroll
    for (int r = 0; r < 8; ++r) {
      int q = qblk + wv * 16 + 8 * half + r;
      Out[((size_t)b * SLEN + q) * DMODEL + d] = outAcc[nt][r];
    }
  }
}

// ---------------------------------------------------------------------------
extern "C" void kernel_launch(void* const* d_in, const int* in_sizes, int n_in,
                              void* d_out, int out_size, void* d_ws, size_t ws_size,
                              hipStream_t stream) {
  (void)in_sizes; (void)n_in; (void)out_size; (void)ws_size;
  const float* hs = (const float*)d_in[0];  // [N,B,S,D]
  const float* Wq = (const float*)d_in[1];
  const float* bq = (const float*)d_in[2];
  const float* Wk = (const float*)d_in[3];
  const float* bk = (const float*)d_in[4];
  const float* Wv = (const float*)d_in[5];
  const float* bv = (const float*)d_in[6];
  float* out = (float*)d_out;

  const size_t qElems  = (size_t)BATCH * SLEN * DMODEL;         // 3.1M
  const size_t kvElems = (size_t)NSRC * BATCH * SLEN * DMODEL;  // 15.7M
  __bf16* Qb = (__bf16*)d_ws;
  __bf16* Kb = Qb + qElems;
  __bf16* Vb = Kb + kvElems;  // total ~69 MB bf16 in workspace

  const int qRows  = BATCH * SLEN;          // 4096
  const int kvRows = NSRC * BATCH * SLEN;   // 20480

  proj_gemm_kernel<<<dim3(qRows / 64, HEADS), 128, 0, stream>>>(
      hs, Wq, bq, Qb, 0.125f);  // 1/sqrt(DH) folded into Q
  proj_gemm_kernel<<<dim3(kvRows / 64, HEADS), 128, 0, stream>>>(
      hs, Wk, bk, Kb, 1.0f);
  proj_gemm_kernel<<<dim3(kvRows / 64, HEADS), 128, 0, stream>>>(
      hs, Wv, bv, Vb, 1.0f);
  flash_attn_kernel<<<dim3(SLEN / 64, HEADS, BATCH), 128, 0, stream>>>(
      Qb, Kb, Vb, out);
}
